// MultiHeadSelfAttention_69243462746780
// MI455X (gfx1250) — matrix operand order repaired and checked by reading, not compiled, not run on hardware
//
#include <hip/hip_runtime.h>
#include <hip/hip_bf16.h>

typedef __attribute__((ext_vector_type(16))) _Float16 v16h;
typedef __attribute__((ext_vector_type(8)))  _Float16 v8h;
typedef __attribute__((ext_vector_type(8)))  float    v8f;

#define BATCH 2
#define SEQ   2048
#define EMB   1024
#define HEADS 16
#define HDIM  64
#define M_TOT (BATCH * SEQ)       // 4096
#define QKVO  (3 * EMB)           // 3072

static __device__ __forceinline__ v8f wmma_f16(v16h a, v16h b, v8f c) {
  return __builtin_amdgcn_wmma_f32_16x16x32_f16(false, a, false, b, (short)0, c, false, false);
}

// A fragment (16x32 f16): row-major source, base pre-offset to (row0, k0).
// Lane l: row = l&15; VGPR0-3 hold k = kb..kb+7, VGPR4-7 hold k = kb+16..kb+23, kb = (l>>4)*8.
static __device__ __forceinline__ v16h load_a16(const _Float16* base, int ld, int lane) {
  const _Float16* p = base + (size_t)(lane & 15) * ld + ((lane >> 4) * 8);
  v8h lo = *(const v8h*)p;
  v8h hi = *(const v8h*)(p + 16);
  v16h a;
#pragma unroll
  for (int i = 0; i < 8; ++i) { a[i] = lo[i]; a[i + 8] = hi[i]; }
  return a;
}

// B fragment (32x16 f16): source stored "col-major" [col][k] (contiguous k), base pre-offset
// to (col0*ld + k0). Lane l: col = l&15; elems 0-7 k = kb..kb+7, elems 8-15 k = kb+16..kb+23, kb = (l>>4)*8.
static __device__ __forceinline__ v16h load_b16(const _Float16* base, int ld, int lane) {
  const _Float16* p = base + (size_t)(lane & 15) * ld + ((lane >> 4) * 8);
  v8h lo = *(const v8h*)p;
  v8h hi = *(const v8h*)(p + 16);
  v16h b;
#pragma unroll
  for (int i = 0; i < 8; ++i) { b[i] = lo[i]; b[i + 8] = hi[i]; }
  return b;
}

// ---------------- prep kernels ----------------
__global__ __launch_bounds__(256) void k_cvt_f16(const float* __restrict__ src,
                                                 _Float16* __restrict__ dst, int n) {
  int i = blockIdx.x * 256 + threadIdx.x;
  if (i < n) dst[i] = (_Float16)src[i];
}

// w: [rows_in][cols_out] f32  ->  wt: [cols_out][rows_in] f16
__global__ __launch_bounds__(256) void k_transpose_f16(const float* __restrict__ w,
                                                       _Float16* __restrict__ wt,
                                                       int rows_in, int cols_out) {
  int i = blockIdx.x * 256 + threadIdx.x;
  if (i < rows_in * cols_out) {
    int o = i / rows_in;
    int r = i - o * rows_in;
    wt[(size_t)o * rows_in + r] = (_Float16)w[(size_t)r * cols_out + o];
  }
}

// ---------------- QKV GEMM: [4096,1024] x [1024,3072] + bias ----------------
// Each wave: 32(M) x 64(O) tile, ping-pong double-buffered K loop (no reg copies).
// Q,K -> [b,h,n,d] f16 ; V -> transposed [b,h,d,n] f16.
__global__ __launch_bounds__(256) void k_qkv(const _Float16* __restrict__ xh,
                                             const _Float16* __restrict__ wqkvT,
                                             const float* __restrict__ bias_qkv,
                                             _Float16* __restrict__ qbuf,
                                             _Float16* __restrict__ kbuf,
                                             _Float16* __restrict__ vtbuf) {
  const int lane = threadIdx.x & 31;
  const int wave = threadIdx.x >> 5;
  const int job = blockIdx.x * 8 + wave;           // 6144 jobs
  const int OSTRIPS = QKVO / 64;                   // 48
  const int ostrip = job % OSTRIPS;
  const int mtile  = job / OSTRIPS;                // 0..127
  const int row0 = mtile * 32;
  const int o0   = ostrip * 64;

  const _Float16* Abase = xh + (size_t)row0 * EMB;
  const _Float16* Bbase = wqkvT + (size_t)o0 * EMB;

  v8f acc[2][4] = {};
  v16h aA[2], bA[4], aB[2], bB[4];

  auto load_set = [&](v16h (&a)[2], v16h (&b)[4], int k0) {
#pragma unroll
    for (int i = 0; i < 2; ++i) a[i] = load_a16(Abase + (size_t)(16 * i) * EMB + k0, EMB, lane);
#pragma unroll
    for (int j = 0; j < 4; ++j) b[j] = load_b16(Bbase + (size_t)(16 * j) * EMB + k0, EMB, lane);
  };
  auto mma_set = [&](const v16h (&a)[2], const v16h (&b)[4]) {
#pragma unroll
    for (int i = 0; i < 2; ++i)
#pragma unroll
      for (int j = 0; j < 4; ++j) acc[i][j] = wmma_f16(a[i], b[j], acc[i][j]);
  };

  load_set(aA, bA, 0);
  for (int k0 = 0; k0 < EMB; k0 += 64) {
    load_set(aB, bB, k0 + 32);          // always in range (EMB multiple of 64)
    mma_set(aA, bA);
    if (k0 + 64 < EMB) load_set(aA, bA, k0 + 64);
    mma_set(aB, bB);
  }

  const int col  = lane & 15;
  const int rofs = (lane >> 4) * 8;
  const int bb   = row0 >> 11;                     // batch (uniform: 32 | 2048)
#pragma unroll
  for (int j = 0; j < 4; ++j) {
    const int o = o0 + 16 * j;                     // uniform
    const float bv = bias_qkv[o + col];
    const int which = o >> 10;                     // 0=Q 1=K 2=V
    const int e = o & (EMB - 1);
    const int h = e >> 6;
    const int dbase = e & (HDIM - 1);
    const int bh = bb * HEADS + h;
#pragma unroll
    for (int i = 0; i < 2; ++i) {
      const int n0 = (row0 + 16 * i + rofs) & (SEQ - 1);
      if (which == 2) {
        v8h vv;
#pragma unroll
        for (int r = 0; r < 8; ++r) vv[r] = (_Float16)(acc[i][j][r] + bv);
        *(v8h*)(vtbuf + ((size_t)bh * HDIM + dbase + col) * SEQ + n0) = vv;
      } else {
        _Float16* dst = (which == 0) ? qbuf : kbuf;
#pragma unroll
        for (int r = 0; r < 8; ++r)
          dst[((size_t)bh * SEQ + (n0 + r)) * HDIM + dbase + col] = (_Float16)(acc[i][j][r] + bv);
      }
    }
  }
}

// ---------------- flash attention: wave = (b,h, 16-query tile) ----------------
__global__ __launch_bounds__(256) void k_attn(const _Float16* __restrict__ qbuf,
                                              const _Float16* __restrict__ kbuf,
                                              const _Float16* __restrict__ vtbuf,
                                              _Float16* __restrict__ oh) {
  __shared__ __align__(16) _Float16 pl[8][16 * 32];
  const int lane = threadIdx.x & 31;
  const int wave = threadIdx.x >> 5;
  const int job = blockIdx.x * 8 + wave;           // 4096 jobs
  const int qtile = job & 127;
  const int bh = job >> 7;                         // 0..31
  const int bb = bh >> 4;
  const int h  = bh & (HEADS - 1);

  const _Float16* Q  = qbuf + ((size_t)bh * SEQ + qtile * 16) * HDIM;
  const _Float16* Kb = kbuf + (size_t)bh * SEQ * HDIM;
  const _Float16* Vt = vtbuf + (size_t)bh * HDIM * SEQ;

  const v16h qa0 = load_a16(Q, HDIM, lane);        // k(d) = 0..31
  const v16h qa1 = load_a16(Q + 32, HDIM, lane);   // k(d) = 32..63

  v8f oacc[4] = {};
  v8f M, L;
#pragma unroll
  for (int r = 0; r < 8; ++r) { M[r] = -3.0e38f; L[r] = 0.0f; }
  const float SC = 0.125f * 1.44269504088896340736f; // D^-0.5 * log2(e)

  auto load_kset = [&](v16h (&kf)[4], int kb) {
    const _Float16* p = Kb + (size_t)kb * HDIM;
    kf[0] = load_b16(p, HDIM, lane);
    kf[1] = load_b16(p + 32, HDIM, lane);
    kf[2] = load_b16(p + (size_t)16 * HDIM, HDIM, lane);
    kf[3] = load_b16(p + (size_t)16 * HDIM + 32, HDIM, lane);
  };

  // one 32-key flash step: scores with kf, prefetch next K block into kfn
  auto step = [&](int kb0, const v16h (&kf)[4], v16h (&kfn)[4]) {
    v8f s0 = {}, s1 = {};
    s0 = wmma_f16(qa0, kf[0], s0);
    s0 = wmma_f16(qa1, kf[1], s0);
    s1 = wmma_f16(qa0, kf[2], s1);
    s1 = wmma_f16(qa1, kf[3], s1);

    // issue V loads (this block) and next-block K loads early: latency hides
    // behind softmax VALU + LDS transpose
    v16h vf[4];
#pragma unroll
    for (int j = 0; j < 4; ++j)
      vf[j] = load_b16(Vt + (size_t)(16 * j) * SEQ + kb0, SEQ, lane);
    if (kb0 + 32 < SEQ) load_kset(kfn, kb0 + 32);

    // online softmax (log2 domain)
    v8f mnew;
#pragma unroll
    for (int r = 0; r < 8; ++r) {
      float a = s0[r] * SC, c = s1[r] * SC;
      s0[r] = a; s1[r] = c;
      mnew[r] = fmaxf(M[r], fmaxf(a, c));
    }
#pragma unroll
    for (int off = 1; off < 16; off <<= 1)
#pragma unroll
      for (int r = 0; r < 8; ++r)
        mnew[r] = fmaxf(mnew[r], __shfl_xor(mnew[r], off, 32));

    v8f rs;
#pragma unroll
    for (int r = 0; r < 8; ++r) {
      float p0 = __builtin_exp2f(s0[r] - mnew[r]);
      float p1 = __builtin_exp2f(s1[r] - mnew[r]);
      s0[r] = p0; s1[r] = p1;
      rs[r] = p0 + p1;
    }
#pragma unroll
    for (int off = 1; off < 16; off <<= 1)
#pragma unroll
      for (int r = 0; r < 8; ++r)
        rs[r] += __shfl_xor(rs[r], off, 32);

#pragma unroll
    for (int r = 0; r < 8; ++r) {
      float alpha = __builtin_exp2f(M[r] - mnew[r]);
      L[r] = L[r] * alpha + rs[r];
      M[r] = mnew[r];
      oacc[0][r] *= alpha; oacc[1][r] *= alpha;
      oacc[2][r] *= alpha; oacc[3][r] *= alpha;
    }

    // P (C layout) -> LDS row-major [16][32] -> A fragment
    {
      const int cc = lane & 15;
      const int mofs = (lane >> 4) * 8;
#pragma unroll
      for (int r = 0; r < 8; ++r) {
        pl[wave][(mofs + r) * 32 + cc]      = (_Float16)s0[r];
        pl[wave][(mofs + r) * 32 + 16 + cc] = (_Float16)s1[r];
      }
    }
    asm volatile("s_wait_dscnt 0x0" ::: "memory");
    v16h pa;
    {
      const int mq = lane & 15;
      const int kq = (lane >> 4) * 8;
      v8h lo = *(const v8h*)&pl[wave][mq * 32 + kq];
      v8h hi = *(const v8h*)&pl[wave][mq * 32 + kq + 16];
#pragma unroll
      for (int i = 0; i < 8; ++i) { pa[i] = lo[i]; pa[i + 8] = hi[i]; }
    }

    // O += P @ V  (B from V^T: contiguous keys)
#pragma unroll
    for (int j = 0; j < 4; ++j)
      oacc[j] = wmma_f16(pa, vf[j], oacc[j]);
  };

  v16h kfA[4], kfB[4];
  load_kset(kfA, 0);
  for (int kb0 = 0; kb0 < SEQ; kb0 += 64) {   // ping-pong, no buffer copies
    step(kb0, kfA, kfB);
    step(kb0 + 32, kfB, kfA);
  }

  // normalize and store to Oh [4096][1024] f16
  v8f inv;
#pragma unroll
  for (int r = 0; r < 8; ++r) inv[r] = 1.0f / L[r];
  const int col  = lane & 15;
  const int rofs = (lane >> 4) * 8;
  const size_t mbase = (size_t)bb * SEQ + qtile * 16 + rofs;
#pragma unroll
  for (int j = 0; j < 4; ++j)
#pragma unroll
    for (int r = 0; r < 8; ++r)
      oh[(mbase + r) * EMB + h * HDIM + 16 * j + col] = (_Float16)(oacc[j][r] * inv[r]);
}

// ---------------- output projection: [4096,1024] x [1024,1024] + bias -> f32 ----------------
// Each wave: 32(M) x 64(O) tile, ping-pong double-buffered K loop.
__global__ __launch_bounds__(256) void k_proj(const _Float16* __restrict__ oh,
                                              const _Float16* __restrict__ wprojT,
                                              const float* __restrict__ bias_proj,
                                              float* __restrict__ out) {
  const int lane = threadIdx.x & 31;
  const int wave = threadIdx.x >> 5;
  const int job = blockIdx.x * 8 + wave;           // 2048 jobs
  const int ostrip = job & 15;                     // 16 strips of 64
  const int mtile  = job >> 4;                     // 0..127
  const int row0 = mtile * 32;
  const int o0   = ostrip * 64;

  const _Float16* Abase = oh + (size_t)row0 * EMB;
  const _Float16* Bbase = wprojT + (size_t)o0 * EMB;

  v8f acc[2][4] = {};
  v16h aA[2], bA[4], aB[2], bB[4];

  auto load_set = [&](v16h (&a)[2], v16h (&b)[4], int k0) {
#pragma unroll
    for (int i = 0; i < 2; ++i) a[i] = load_a16(Abase + (size_t)(16 * i) * EMB + k0, EMB, lane);
#pragma unroll
    for (int j = 0; j < 4; ++j) b[j] = load_b16(Bbase + (size_t)(16 * j) * EMB + k0, EMB, lane);
  };
  auto mma_set = [&](const v16h (&a)[2], const v16h (&b)[4]) {
#pragma unroll
    for (int i = 0; i < 2; ++i)
#pragma unroll
      for (int j = 0; j < 4; ++j) acc[i][j] = wmma_f16(a[i], b[j], acc[i][j]);
  };

  load_set(aA, bA, 0);
  for (int k0 = 0; k0 < EMB; k0 += 64) {
    load_set(aB, bB, k0 + 32);
    mma_set(aA, bA);
    if (k0 + 64 < EMB) load_set(aA, bA, k0 + 64);
    mma_set(aB, bB);
  }

  const int col  = lane & 15;
  const int rofs = (lane >> 4) * 8;
#pragma unroll
  for (int j = 0; j < 4; ++j) {
    const int o = o0 + 16 * j;
    const float bv = bias_proj[o + col];
#pragma unroll
    for (int i = 0; i < 2; ++i)
#pragma unroll
      for (int r = 0; r < 8; ++r)
        out[(size_t)(row0 + 16 * i + rofs + r) * EMB + o + col] = acc[i][j][r] + bv;
  }
}

extern "C" void kernel_launch(void* const* d_in, const int* in_sizes, int n_in,
                              void* d_out, int out_size, void* d_ws, size_t ws_size,
                              hipStream_t stream) {
  const float* x      = (const float*)d_in[0];  // [2,2048,1024]
  const float* w_qkv  = (const float*)d_in[1];  // [1024,3072]
  const float* b_qkv  = (const float*)d_in[2];  // [3072]
  const float* w_proj = (const float*)d_in[3];  // [1024,1024]
  const float* b_proj = (const float*)d_in[4];  // [1024]
  float* out = (float*)d_out;                   // [2,2048,1024]

  char* ws = (char*)d_ws;
  const size_t MB = 1ull << 20;
  _Float16* xh     = (_Float16*)(ws + 0);        // 8 MB
  _Float16* wqkvT  = (_Float16*)(ws + 8 * MB);   // 6 MB
  _Float16* wprojT = (_Float16*)(ws + 14 * MB);  // 2 MB
  _Float16* qbuf   = (_Float16*)(ws + 16 * MB);  // 8 MB
  _Float16* kbuf   = (_Float16*)(ws + 24 * MB);  // 8 MB
  _Float16* vtbuf  = (_Float16*)(ws + 32 * MB);  // 8 MB
  _Float16* ohbuf  = (_Float16*)(ws + 40 * MB);  // 8 MB  (total 48 MB)

  // prep: convert + transpose
  k_cvt_f16<<<(M_TOT * EMB + 255) / 256, 256, 0, stream>>>(x, xh, M_TOT * EMB);
  k_transpose_f16<<<(EMB * QKVO + 255) / 256, 256, 0, stream>>>(w_qkv, wqkvT, EMB, QKVO);
  k_transpose_f16<<<(EMB * EMB + 255) / 256, 256, 0, stream>>>(w_proj, wprojT, EMB, EMB);

  // QKV GEMM: 128 mtiles * 48 ostrips = 6144 wave jobs, 8 waves/block
  k_qkv<<<6144 / 8, 256, 0, stream>>>(xh, wqkvT, b_qkv, qbuf, kbuf, vtbuf);

  // attention: 32 (b,h) * 128 qtiles = 4096 wave jobs
  k_attn<<<4096 / 8, 256, 0, stream>>>(qbuf, kbuf, vtbuf, ohbuf);

  // projection: 128 mtiles * 16 ostrips = 2048 wave jobs
  k_proj<<<2048 / 8, 256, 0, stream>>>(ohbuf, wprojT, b_proj, out);
}
